// LSTMCell_49460843380786
// MI455X (gfx1250) — matrix operand order repaired and checked by reading, not compiled, not run on hardware
//
#include <hip/hip_runtime.h>

// ---------------------------------------------------------------------------
// LSTM cell for MI455X (gfx1250, wave32, WMMA).
// gates = X @ Wih^T + H @ Whh^T + b ; fused sigmoid/tanh epilogue.
// bf16 WMMA (16x16x32) with f32 accumulation; fp32->bf16 conversion happens
// once per staged LDS tile. Compute-bound (AI ~600 FLOP/B); working set fits
// in 192MB L2 so tile re-reads are L2 hits, not HBM traffic.
//
// v2: wave M-tile raised 32 -> 64 (4 M-subtiles x 4 gates = 16 WMMA per
// K-step per wave) so the LDS-read : WMMA ratio drops from 1.5 to 1.0
// ds_load_b128 per WMMA, and staging/barrier cost is amortized over 2x FLOPs.
// ---------------------------------------------------------------------------

typedef __attribute__((ext_vector_type(16))) __bf16 v16bf;
typedef __attribute__((ext_vector_type(8)))  __bf16 bf16x8;
typedef __attribute__((ext_vector_type(4)))  __bf16 bf16x4;
typedef __attribute__((ext_vector_type(8)))  float  v8f;
typedef __attribute__((ext_vector_type(4)))  float  f32x4;

namespace {
constexpr int kB   = 4096;  // batch (GEMM M)
constexpr int kH   = 1024;  // hidden (and input) size
constexpr int kK   = 1024;  // reduction dim
constexpr int MBLK = 256;   // rows per block (4 waves x 64)
constexpr int KT   = 32;    // K per stage (one bf16 WMMA K)
constexpr int LDSW = 40;    // LDS row stride in bf16 (32 + 8 pad -> no conflicts)
}

__device__ __forceinline__ float fast_sigmoid(float x) {
  return 1.0f / (1.0f + __expf(-x));
}
__device__ __forceinline__ float fast_tanh(float x) {
  x = fminf(15.0f, fmaxf(-15.0f, x));
  float e = __expf(2.0f * x);
  return (e - 1.0f) / (e + 1.0f);
}

__global__ __launch_bounds__(256)
void lstm_cell_wmma_bf16(const float* __restrict__ X,
                         const float* __restrict__ Hin,
                         const float* __restrict__ Cin,
                         const float* __restrict__ Wih,
                         const float* __restrict__ Whh,
                         const float* __restrict__ Bih,
                         const float* __restrict__ Bhh,
                         float* __restrict__ Hout,
                         float* __restrict__ Cout) {
  __shared__ __bf16 As[MBLK * LDSW];  // 256 x 32 activation tile (bf16)
  __shared__ __bf16 Bs[128 * LDSW];   // 128 W rows = 4 gates x 32 h-cols (bf16)

  const int tid    = threadIdx.x;
  const int lane   = tid & 31;
  const int wave   = tid >> 5;
  const int wave_m = wave & 3;   // 4 waves along M, 64 rows each
  const int wave_h = wave >> 2;  // 2 waves along h, 16 cols each

  const int mblock = blockIdx.x; // 16 blocks * 256 rows = 4096
  const int hblock = blockIdx.y; // 32 blocks * 32 h-cols = 1024

  // Staging decomposition: 256 threads, one float4 per row per thread,
  // 8 threads cover a 32-col row; 32 rows per pass.
  const int st_c = (tid & 7) * 4;  // column 0,4,...,28
  const int st_r = tid >> 3;       // row 0..31 (+32*j)

  v8f acc[4][4];
#pragma unroll
  for (int mi = 0; mi < 4; ++mi)
#pragma unroll
    for (int g = 0; g < 4; ++g)
#pragma unroll
      for (int v = 0; v < 8; ++v) acc[mi][g][v] = 0.0f;

  // Per-lane fragment addressing (contiguous K order, same for A and B, wave32):
  // A 16x32: lanes 0-15 hold K=0..15, lanes 16-31 hold K=16..31 (M = lane%16).
  // B 32x16: lanes 0-15 hold K=0..15, lanes 16-31 hold K=16..31 (N = lane%16).
  const int arow = lane & 15;
  const int akb  = (lane < 16) ? 0 : 16;  // A K-base (elements)
  const int bkb  = (lane < 16) ? 0 : 16;  // B K-base (elements)

  union Frag { v16bf v; bf16x8 p[2]; };

#pragma unroll 1
  for (int pass = 0; pass < 2; ++pass) {
    const float* Ag = pass ? Hin : X;    // activations
    const float* Wg = pass ? Whh : Wih;  // stacked [4H, K] weights

    // Row pointers for staging (advance by KT each K step).
    const float* apj[8];
    const float* bpj[4];
#pragma unroll
    for (int j = 0; j < 8; ++j)
      apj[j] = Ag + (size_t)(mblock * MBLK + st_r + 32 * j) * kK + st_c;
#pragma unroll
    for (int j = 0; j < 4; ++j)  // B tile row (j*32+st_r) <-> W row j*1024+...
      bpj[j] = Wg + (size_t)(j * kH + hblock * 32 + st_r) * kK + st_c;

#pragma unroll 1
    for (int k0 = 0; k0 < kK; k0 += KT) {
      // Issue global loads before the barrier so they overlap prior compute.
      f32x4 areg[8], breg[4];
#pragma unroll
      for (int j = 0; j < 8; ++j) {
        areg[j] = *(const f32x4*)apj[j];
        apj[j] += KT;
      }
#pragma unroll
      for (int j = 0; j < 4; ++j) {
        breg[j] = *(const f32x4*)bpj[j];
        bpj[j] += KT;
      }

      __syncthreads();  // all waves done reading previous LDS tile
#pragma unroll
      for (int j = 0; j < 8; ++j) {
        bf16x4 a4;
#pragma unroll
        for (int t = 0; t < 4; ++t) a4[t] = (__bf16)areg[j][t];
        *(bf16x4*)&As[(st_r + 32 * j) * LDSW + st_c] = a4;
      }
#pragma unroll
      for (int j = 0; j < 4; ++j) {
        bf16x4 b4;
#pragma unroll
        for (int t = 0; t < 4; ++t) b4[t] = (__bf16)breg[j][t];
        *(bf16x4*)&Bs[(st_r + 32 * j) * LDSW + st_c] = b4;
      }
      __syncthreads();  // tile visible to all waves

      // Prefetch all four gate B-fragments (reused by 4 M-subtiles).
      Frag bfr[4];
#pragma unroll
      for (int g = 0; g < 4; ++g) {
        const int bb = (g * 32 + wave_h * 16 + arow) * LDSW + bkb;
        bfr[g].p[0] = *(const bf16x8*)&Bs[bb];        // K = bkb..bkb+7
        bfr[g].p[1] = *(const bf16x8*)&Bs[bb + 8];    // K = bkb+8..bkb+15
      }

#pragma unroll
      for (int mi = 0; mi < 4; ++mi) {
        Frag af;
        const int ab = (wave_m * 64 + mi * 16 + arow) * LDSW + akb;
        af.p[0] = *(const bf16x8*)&As[ab];        // K = akb..akb+7
        af.p[1] = *(const bf16x8*)&As[ab + 8];    // K = akb+8..akb+15
#pragma unroll
        for (int g = 0; g < 4; ++g)
          acc[mi][g] = __builtin_amdgcn_wmma_f32_16x16x32_bf16(
              false, af.v, false, bfr[g].v, (short)0, acc[mi][g],
              false, false);
      }
    }
  }

  // -------- fused LSTM epilogue (all 4 gates resident in this wave) --------
  // C/D layout: lanes 0-15 -> N=lane, M=v ; lanes 16-31 -> N=lane-16, M=v+8.
  const int hidx = hblock * 32 + wave_h * 16 + (lane & 15);
  float bsum[4];
#pragma unroll
  for (int g = 0; g < 4; ++g)
    bsum[g] = Bih[g * kH + hidx] + Bhh[g * kH + hidx];

#pragma unroll
  for (int mi = 0; mi < 4; ++mi) {
    const int mbase =
        mblock * MBLK + wave_m * 64 + mi * 16 + ((lane < 16) ? 0 : 8);
#pragma unroll
    for (int v = 0; v < 8; ++v) {
      const size_t idx = (size_t)(mbase + v) * kH + hidx;
      const float ig = fast_sigmoid(acc[mi][0][v] + bsum[0]);
      const float fg = fast_sigmoid(acc[mi][1][v] + bsum[1]);
      const float gg = fast_tanh   (acc[mi][2][v] + bsum[2]);
      const float og = fast_sigmoid(acc[mi][3][v] + bsum[3]);
      const float cn = fg * Cin[idx] + ig * gg;
      Cout[idx] = cn;
      Hout[idx] = og * fast_tanh(cn);
    }
  }
}

extern "C" void kernel_launch(void* const* d_in, const int* in_sizes, int n_in,
                              void* d_out, int out_size, void* d_ws,
                              size_t ws_size, hipStream_t stream) {
  (void)in_sizes; (void)n_in; (void)out_size; (void)d_ws; (void)ws_size;
  const float* X   = (const float*)d_in[0];
  const float* Hin = (const float*)d_in[1];
  const float* Cin = (const float*)d_in[2];
  const float* Wih = (const float*)d_in[3];
  const float* Whh = (const float*)d_in[4];
  const float* Bih = (const float*)d_in[5];
  const float* Bhh = (const float*)d_in[6];

  float* out  = (float*)d_out;
  float* Hout = out;                            // h' first in tuple order
  float* Cout = out + (size_t)kB * kH;          // then c'

  dim3 grid(kB / MBLK, kH / 32);                // 16 x 32 blocks
  dim3 block(256);                              // 8 waves (wave32)
  lstm_cell_wmma_bf16<<<grid, block, 0, stream>>>(X, Hin, Cin, Wih, Whh,
                                                  Bih, Bhh, Hout, Cout);
}
